// RoleExtractor_2851858284998
// MI455X (gfx1250) — matrix operand order repaired and checked by reading, not compiled, not run on hardware
//
#include <hip/hip_runtime.h>
#include <hip/hip_bf16.h>

#define ALPHA 0.15f

typedef __attribute__((ext_vector_type(16))) _Float16 v16h;
typedef __attribute__((ext_vector_type(8)))  float    v8f;

// ---------------- small setup kernels ----------------

__global__ void init_deg_pr_kernel(float* __restrict__ deg, float* __restrict__ pr, int n) {
    int i = blockIdx.x * blockDim.x + threadIdx.x;
    if (i < n) { deg[i] = 0.0f; pr[i] = 1.0f / (float)n; }
}

// 4 edges per thread: int4 index loads -> 4 atomics
__global__ void deg_kernel(const int* __restrict__ src, float* __restrict__ deg, int e) {
    int base = (blockIdx.x * blockDim.x + threadIdx.x) * 4;
    if (base + 3 < e) {
        int4 s = *(const int4*)(src + base);
        atomicAdd(&deg[s.x], 1.0f);
        atomicAdd(&deg[s.y], 1.0f);
        atomicAdd(&deg[s.z], 1.0f);
        atomicAdd(&deg[s.w], 1.0f);
    } else {
        for (int i = base; i < e; ++i) atomicAdd(&deg[src[i]], 1.0f);
    }
}

__global__ void invdeg_kernel(const float* __restrict__ deg, float* __restrict__ inv, int n) {
    int i = blockIdx.x * blockDim.x + threadIdx.x;
    if (i < n) inv[i] = 1.0f / fmaxf(deg[i], 1.0f);
}

__global__ void fill_kernel(float* __restrict__ p, float v, int n) {
    int i = blockIdx.x * blockDim.x + threadIdx.x;
    if (i < n) p[i] = v;
}

// 4 edges per thread: two b128 index loads, 4 gathers, 4 atomics.
// Edge list (51MB) + pr (400KB) are L2-resident on MI455X (192MB L2).
__global__ void scatter_kernel(const int* __restrict__ src, const int* __restrict__ dst,
                               const float* __restrict__ pr, const float* __restrict__ inv,
                               float* __restrict__ npr, int e) {
    int base = (blockIdx.x * blockDim.x + threadIdx.x) * 4;
    if (base + 3 < e) {
        int4 s = *(const int4*)(src + base);
        int4 d = *(const int4*)(dst + base);
        float c0 = (1.0f - ALPHA) * pr[s.x] * inv[s.x];
        float c1 = (1.0f - ALPHA) * pr[s.y] * inv[s.y];
        float c2 = (1.0f - ALPHA) * pr[s.z] * inv[s.z];
        float c3 = (1.0f - ALPHA) * pr[s.w] * inv[s.w];
        atomicAdd(&npr[d.x], c0);
        atomicAdd(&npr[d.y], c1);
        atomicAdd(&npr[d.z], c2);
        atomicAdd(&npr[d.w], c3);
    } else {
        for (int i = base; i < e; ++i) {
            int s = src[i];
            atomicAdd(&npr[dst[i]], (1.0f - ALPHA) * pr[s] * inv[s]);
        }
    }
}

// M = G^{-1} A where A = behavior_matrix^T [4x8], G = A A^T [4x4].
// Row 1 of M pre-divided by N (pagerank column of X is pre-multiplied by N
// before the f16 WMMA so its values stay in f16's normal range).
__global__ void compute_M_kernel(const float* __restrict__ bm, float* __restrict__ M, float n) {
    if (threadIdx.x != 0 || blockIdx.x != 0) return;
    float A[4][8];
    for (int r = 0; r < 8; ++r)
        for (int k = 0; k < 4; ++k)
            A[k][r] = bm[r * 4 + k];
    float G[4][8]; // augmented [G | I]
    for (int i = 0; i < 4; ++i) {
        for (int j = 0; j < 4; ++j) {
            float s = 0.0f;
            for (int r = 0; r < 8; ++r) s += A[i][r] * A[j][r];
            G[i][j] = s;
            G[i][4 + j] = (i == j) ? 1.0f : 0.0f;
        }
    }
    for (int c = 0; c < 4; ++c) {                 // Gauss-Jordan w/ partial pivot
        int p = c;
        for (int rr = c + 1; rr < 4; ++rr)
            if (fabsf(G[rr][c]) > fabsf(G[p][c])) p = rr;
        if (p != c)
            for (int j = 0; j < 8; ++j) { float t = G[c][j]; G[c][j] = G[p][j]; G[p][j] = t; }
        float ip = 1.0f / G[c][c];
        for (int j = 0; j < 8; ++j) G[c][j] *= ip;
        for (int rr = 0; rr < 4; ++rr) {
            if (rr == c) continue;
            float f = G[rr][c];
            for (int j = 0; j < 8; ++j) G[rr][j] -= f * G[c][j];
        }
    }
    for (int k = 0; k < 4; ++k)
        for (int r = 0; r < 8; ++r) {
            float s = 0.0f;
            for (int j = 0; j < 4; ++j) s += G[k][4 + j] * A[j][r];
            if (k == 1) s /= n;                    // undo pagerank pre-scaling
            M[k * 8 + r] = s;
        }
}

// ---------------- fused WMMA kernel: emb2 GEMM + X@M + normalize ----------------
// 128 threads = 4 waves; each wave owns one 16-row tile.
__global__ __launch_bounds__(128) void roles_kernel(
    const float* __restrict__ emb,    // [N,128]
    const float* __restrict__ rw,     // [128,2]
    const float* __restrict__ rb,     // [2]
    const int*   __restrict__ batch,  // [N]
    const float* __restrict__ deg,    // [N]
    const float* __restrict__ pr,     // [N]
    const float* __restrict__ M,      // [4,8]
    float* __restrict__ out,          // [N,8]
    int n)
{
    // B operand in fragment-friendly layout, cols 2..15 pre-zeroed:
    // s_bw[col][k] = resizer_W[k][col] (half). Lane fragment = 16 contiguous halfs.
    __shared__ _Float16 s_bw[16][128];
    __shared__ float    s_M[32];          // M [4][8]
    __shared__ float    s_e2[4][16][2];   // per-wave emb2 tile (row-major)

    const int tid      = threadIdx.x;
    const int lane     = tid & 31;
    const int wave     = tid >> 5;
    const int row      = lane & 15;       // A row / D col
    const int half_sel = lane >> 4;       // 0: lanes 0-15, 1: lanes 16-31
    const int col      = row;             // B/D column for this lane

    const int ntiles = (n + 15) >> 4;
    int tile = blockIdx.x * 4 + wave;
    if (tile >= ntiles) tile = ntiles - 1;   // tail waves redo last tile (benign)

    // ---- cooperative staging (branch-free, vectorized) ----
    {   // zero padding rows 2..15: 3584 B = 224 float4
        float4* z = (float4*)&s_bw[2][0];
        float4 zv = make_float4(0.0f, 0.0f, 0.0f, 0.0f);
        if (tid < 224) z[tid] = zv;
        if (tid + 128 < 224) z[tid + 128] = zv;
        // rows 0,1: one float2 of resizer_W per thread (k = tid)
        float2 wv = ((const float2*)rw)[tid];
        s_bw[0][tid] = (_Float16)wv.x;
        s_bw[1][tid] = (_Float16)wv.y;
        if (tid < 32) s_M[tid] = M[tid];
    }
    __syncthreads();

    int g_row = tile * 16 + row;
    int gr    = g_row < n ? g_row : (n - 1);   // clamp loads; EXEC stays full for WMMA
    const float* arow = emb + (size_t)gr * 128;

    // ---- emb2 tile: 4x v_wmma_f32_16x16x32_f16 over K=128 ----
    v8f acc = {};
    #pragma unroll
    for (int kc = 0; kc < 4; ++kc) {
        // A fragment: halfs 0..15 = K kbase..+15 (contiguous)
        int kbase = kc * 32 + half_sel * 16;
        v16h a;
        #pragma unroll
        for (int i = 0; i < 8; ++i) {
            a[i]     = (_Float16)arow[kbase + i];
            a[8 + i] = (_Float16)arow[kbase + 8 + i];
        }
        // B fragment: 16 contiguous pre-padded halfs -> plain vector LDS load
        int kb = kc * 32 + half_sel * 16;
        v16h b = *(const v16h*)&s_bw[col][kb];
        acc = __builtin_amdgcn_wmma_f32_16x16x32_f16(false, a, false, b, (short)0, acc,
                                                     false, false);
    }

    // D layout: lane col = lane&15, VGPR j -> row m = j + half_sel*8.
    // Transpose the two valid columns (+bias) through LDS into row-major.
    if (col < 2) {
        float bias = rb[col];
        #pragma unroll
        for (int j = 0; j < 8; ++j)
            s_e2[wave][j + half_sel * 8][col] = acc[j] + bias;
    }
    __syncthreads();

    // ---- W = X @ M via one WMMA (K padded 4 -> 32), branch-free fragments ----
    const float msk = (half_sel == 0) ? 1.0f : 0.0f;   // lanes 16-31: zero K rows
    int   bidx = batch[gr];
    float dv = deg[bidx];
    float pv = pr[bidx] * (float)n;                    // pre-scaled; M row 1 compensates
    float e0 = s_e2[wave][row][0];
    float e1 = s_e2[wave][row][1];
    v16h xa = {};
    xa[0] = (_Float16)(dv * msk);
    xa[1] = (_Float16)(pv * msk);
    xa[2] = (_Float16)(e0 * msk);
    xa[3] = (_Float16)(e1 * msk);

    const float msk2 = (half_sel == 0 && col < 8) ? 1.0f : 0.0f;
    const int   cc2  = col & 7;
    v16h xb = {};
    #pragma unroll
    for (int i = 0; i < 4; ++i)
        xb[i] = (_Float16)(s_M[i * 8 + cc2] * msk2);

    v8f w = {};
    w = __builtin_amdgcn_wmma_f32_16x16x32_f16(false, xa, false, xb, (short)0, w,
                                               false, false);

    // Row L2-normalize + store. Row m's 8 values live in lanes {0..7} (m<8) or
    // {16..23} (m>=8) at VGPR m&7 -> xor-shuffle 1,2,4 reduces within the row.
    const bool do_store = (col < 8);
    #pragma unroll
    for (int j = 0; j < 8; ++j) {
        float v = w[j];
        float s = v * v;
        s += __shfl_xor(s, 1, 32);
        s += __shfl_xor(s, 2, 32);
        s += __shfl_xor(s, 4, 32);
        float r = v / fmaxf(sqrtf(s), 1e-12f);
        int m  = j + half_sel * 8;
        int go = tile * 16 + m;
        if (do_store && go < n)
            out[(size_t)go * 8 + col] = r;
    }
}

// ---------------- launch ----------------

extern "C" void kernel_launch(void* const* d_in, const int* in_sizes, int n_in,
                              void* d_out, int out_size, void* d_ws, size_t ws_size,
                              hipStream_t stream) {
    const int*   edge  = (const int*)d_in[0];     // [2,E]
    const int    E     = in_sizes[0] / 2;
    const int*   src   = edge;
    const int*   dst   = edge + E;
    const int*   batch = (const int*)d_in[1];     // [N]
    const int    n     = in_sizes[1];
    const float* emb   = (const float*)d_in[2];   // [N,128]
    const float* rw    = (const float*)d_in[3];   // [128,2]
    const float* rb    = (const float*)d_in[4];   // [2]
    const float* bm    = (const float*)d_in[5];   // [8,4]

    float* ws   = (float*)d_ws;
    float* deg  = ws;
    float* inv  = ws + (size_t)n;
    float* pra  = ws + (size_t)2 * n;
    float* prb  = ws + (size_t)3 * n;
    float* M    = ws + (size_t)4 * n;

    const int TB = 256;
    const int nb_n  = (n + TB - 1) / TB;
    const int E4    = (E + 3) / 4;                 // threads for 4-edge kernels
    const int nb_e4 = (E4 + TB - 1) / TB;

    init_deg_pr_kernel<<<nb_n, TB, 0, stream>>>(deg, pra, n);
    compute_M_kernel<<<1, 32, 0, stream>>>(bm, M, (float)n);
    deg_kernel<<<nb_e4, TB, 0, stream>>>(src, deg, E);
    invdeg_kernel<<<nb_n, TB, 0, stream>>>(deg, inv, n);

    float* cur = pra;
    float* nxt = prb;
    for (int it = 0; it < 10; ++it) {
        fill_kernel<<<nb_n, TB, 0, stream>>>(nxt, ALPHA / (float)n, n);
        scatter_kernel<<<nb_e4, TB, 0, stream>>>(src, dst, cur, inv, nxt, E);
        float* t = cur; cur = nxt; nxt = t;
    }

    const int ntiles = (n + 15) / 16;
    roles_kernel<<<(ntiles + 3) / 4, 128, 0, stream>>>(emb, rw, rb, batch, deg, cur, M,
                                                       (float*)d_out, n);
}
